// AttentionWithKVCache_69501160784052
// MI455X (gfx1250) — matrix operand order repaired and checked by reading, not compiled, not run on hardware
//
#include <hip/hip_runtime.h>
#include <math.h>

// Problem constants (fixed by the reference's setup_inputs).
#define H_      32
#define KVH_    8
#define Dh      128
#define DIM_    4096
#define T_      256
#define MAXSEQ_ 8192

typedef __attribute__((ext_vector_type(16))) _Float16     v16h;
typedef __attribute__((ext_vector_type(8)))  float        v8f;
typedef __attribute__((ext_vector_type(4)))  float        v4f;
typedef __attribute__((ext_vector_type(4)))  unsigned int v4u;

// Scheduler pipeline hints (masks: 0x8=WMMA, 0x20=VMEM read, 0x100=DS read).
#if __has_builtin(__builtin_amdgcn_sched_group_barrier)
#define SGB(mask, cnt, id) __builtin_amdgcn_sched_group_barrier((mask), (cnt), (id))
#else
#define SGB(mask, cnt, id)
#endif

__device__ __forceinline__ unsigned pack_h2(float a, float b) {
  union { _Float16 h[2]; unsigned u; } p;
  p.h[0] = (_Float16)a; p.h[1] = (_Float16)b;
  return p.u;
}

// Load 32 contiguous fp32 and pack to 16 dwords of f16 pairs (kept in VGPRs).
__device__ __forceinline__ void pack32(const float* __restrict__ p, unsigned t[16]) {
#pragma unroll
  for (int i = 0; i < 8; ++i) {
    v4f f = *(const v4f*)(p + i * 4);
    t[2*i]   = pack_h2(f[0], f[1]);
    t[2*i+1] = pack_h2(f[2], f[3]);
  }
}
// Store those 16 dwords (32 f16) to LDS as 4x b128.
__device__ __forceinline__ void store32(_Float16* d, const unsigned t[16]) {
#pragma unroll
  for (int i = 0; i < 4; ++i) {
    v4u u = { t[4*i], t[4*i+1], t[4*i+2], t[4*i+3] };
    *(v4u*)(d + i * 8) = u;
  }
}

// A-operand fragment (16x32 f16), source row-major [M][K]:
// lanes 0-15: row r, K = kk+{0..7} (v0-3) and kk+{16..23} (v4-7)
// lanes 16-31: row r, K = kk+{8..15} and kk+{24..31}
__device__ __forceinline__ v16h ld_fragA(const _Float16* tile, int kk, int ldt) {
  const int lane = threadIdx.x & 31;
  const int r = lane & 15, hf = lane >> 4;
  const _Float16* p = tile + r * ldt + kk + hf * 8;
  union { v4u u[2]; v16h h; } q;
  q.u[0] = *(const v4u*)(p);
  q.u[1] = *(const v4u*)(p + 16);
  return q.h;
}

// B-operand fragment (32x16 f16) from row-major [N][K] storage (i.e. B^T rows):
// lane r holds column N=r; lanes 0-15: K = kk+{0..7},{16..23}, lanes 16-31: K = kk+{8..15},{24..31}.
__device__ __forceinline__ v16h ld_fragB(const _Float16* tile, int kk, int ldt) {
  const int lane = threadIdx.x & 31;
  const int r = lane & 15, hf = lane >> 4;
  const _Float16* p = tile + r * ldt + kk + hf * 8;
  union { v4u u[2]; v16h h; } q;
  q.u[0] = *(const v4u*)(p);
  q.u[1] = *(const v4u*)(p + 16);
  return q.h;
}

__device__ __forceinline__ v8f wmma16(v16h a, v16h b, v8f c) {
  // v_wmma_f32_16x16x32_f16 (8-arg form, codegen-verified on gfx1250)
  return __builtin_amdgcn_wmma_f32_16x16x32_f16(
      false, a, false, b, (short)0, c, false, false);
}

// ---------------------------------------------------------------------------
// C[M x N] = A[M x K] @ B[N x K]^T. A,B,C fp32 row-major in HBM; tiles are
// converted to f16 while staging into LDS; accumulate in fp32 via WMMA.
// Block: 256 threads (8 waves), tile 128(M) x 128(N), BK=64.
// Wave w computes rows [16w,16w+16) x all 128 columns (8 accumulators).
// Both A-fragments are hoisted so the DS-read stream is uniform; the
// sched_group_barrier pipeline keeps one B-fragment in flight per WMMA
// (prime 8 reads, steady {1 WMMA, 2 DS reads}, drain 2).
// ---------------------------------------------------------------------------
#define BM  128
#define BN  128
#define BK  64
#define BKP 72   // padded LDS row stride (f16 elems), multiple of 8 for b128

__global__ __launch_bounds__(256) void gemm_nt_kernel(
    const float* __restrict__ A, const float* __restrict__ B,
    float* __restrict__ C, int N, int K)
{
  __shared__ alignas(16) _Float16 As[BM][BKP];
  __shared__ alignas(16) _Float16 Bs[BN][BKP];

  const int tid  = threadIdx.x;
  const int wave = tid >> 5;
  const int lane = tid & 31;
  const int r = lane & 15, hf = lane >> 4;
  const int m0 = blockIdx.y * BM;
  const int n0 = blockIdx.x * BN;

  v8f acc[8] = {};

  // Staging assignment: thread covers 32 contiguous K of one row (A and B).
  const int srow  = tid >> 1;
  const int skoff = (tid & 1) * 32;
  const float* pa = A + (size_t)(m0 + srow) * K + skoff;
  const float* pb = B + (size_t)(n0 + srow) * K + skoff;
  _Float16* da = &As[srow][skoff];
  _Float16* db = &Bs[srow][skoff];

  unsigned ta[16], tb[16];
  pack32(pa, ta);            // preload K-tile 0
  pack32(pb, tb);

  for (int k0 = 0; k0 < K; k0 += BK) {
    store32(da, ta);         // regs -> LDS (current tile)
    store32(db, tb);
    __syncthreads();

    if (k0 + BK < K) {       // issue next tile's global loads before compute
      pack32(pa + k0 + BK, ta);
      pack32(pb + k0 + BK, tb);
      __builtin_prefetch(pa + k0 + 2 * BK, 0, 1);
      __builtin_prefetch(pb + k0 + 2 * BK, 0, 1);
    }

    // Hoist both A-fragments; then a uniform stream of 16 B-frag loads.
    v16h a0 = ld_fragA(&As[wave * 16][0], 0,  BKP);
    v16h a1 = ld_fragA(&As[wave * 16][0], 32, BKP);
#pragma unroll
    for (int n = 0; n < 8; ++n) {
      v16h b = ld_fragB(&Bs[n * 16][0], 0, BKP);
      acc[n] = wmma16(a0, b, acc[n]);
    }
#pragma unroll
    for (int n = 0; n < 8; ++n) {
      v16h b = ld_fragB(&Bs[n * 16][0], 32, BKP);
      acc[n] = wmma16(a1, b, acc[n]);
    }
    // Pipeline: 16 global reads first; prime 8 DS reads (a0,a1,b0,b1),
    // steady {1 WMMA, 2 DS reads} x14, drain 2 WMMAs. (36 reads, 16 WMMAs.)
    SGB(0x020, 16, 0);
    SGB(0x100, 8, 0);
#pragma unroll
    for (int i = 0; i < 14; ++i) { SGB(0x008, 1, 0); SGB(0x100, 2, 0); }
    SGB(0x008, 1, 0);
    SGB(0x008, 1, 0);

    __syncthreads();
  }

  const int mrow = m0 + wave * 16 + hf * 8;  // C layout: M = j + 8*hf, N = r
#pragma unroll
  for (int n = 0; n < 8; ++n)
#pragma unroll
    for (int j = 0; j < 8; ++j)
      C[(size_t)(mrow + j) * N + n0 + n * 16 + r] = acc[n][j];
}

// ---------------------------------------------------------------------------
// RoPE on Q and K; reorder new K/V into [kvh][T][D] extension buffers
// (the input caches are NEVER mutated); Q stored as f16 [h][T][D].
// ---------------------------------------------------------------------------
__global__ __launch_bounds__(256) void rope_kernel(
    const float* __restrict__ Qraw, const float* __restrict__ Kraw,
    const float* __restrict__ Vraw, const float* __restrict__ cosv,
    const float* __restrict__ sinv, _Float16* __restrict__ Qb,
    float* __restrict__ Kext, float* __restrict__ Vext)
{
  const int t = blockIdx.x;
  const float* c = cosv + (size_t)t * Dh;
  const float* s = sinv + (size_t)t * Dh;

  for (int i = threadIdx.x; i < H_ * Dh; i += blockDim.x) {
    int d = i & (Dh - 1);
    int h = i >> 7;
    float q  = Qraw[(size_t)t * (H_ * Dh) + i];
    float qo = Qraw[(size_t)t * (H_ * Dh) + (d < 64 ? i + 64 : i - 64)];
    float rr = q * c[d] + (d < 64 ? -qo : qo) * s[d];
    Qb[((size_t)h * T_ + t) * Dh + d] = (_Float16)rr;
  }
  for (int i = threadIdx.x; i < KVH_ * Dh; i += blockDim.x) {
    int d = i & (Dh - 1);
    int g = i >> 7;
    float k  = Kraw[(size_t)t * (KVH_ * Dh) + i];
    float ko = Kraw[(size_t)t * (KVH_ * Dh) + (d < 64 ? i + 64 : i - 64)];
    Kext[((size_t)g * T_ + t) * Dh + d] = k * c[d] + (d < 64 ? -ko : ko) * s[d];
    Vext[((size_t)g * T_ + t) * Dh + d] = Vraw[(size_t)t * (KVH_ * Dh) + i];
  }
}

// ---------------------------------------------------------------------------
// Flash-style attention. Grid (H, T/128); block = 256 threads = 8 waves.
// Wave w owns query rows [128*blockIdx.y + 16w, +16). S swept in 64-row
// chunks; K chunk staged f16 row-major, V chunk staged f16 TRANSPOSED
// (Vt[d][s]) so P@V B-fragments are contiguous LDS reads. Requires
// (pos + T) % 64 == 0 (4352 here).
// ---------------------------------------------------------------------------
#define SC  64
#define KSP 136
#define VTP 72
#define PSP 72

__global__ __launch_bounds__(256) void attn_kernel(
    const _Float16* __restrict__ Qb, const float* __restrict__ kcache,
    const float* __restrict__ vcache, const float* __restrict__ Kext,
    const float* __restrict__ Vext, const int* __restrict__ posp,
    float* __restrict__ O)
{
  __shared__ alignas(16) _Float16 Ks[SC][KSP];   // [s][d]
  __shared__ alignas(16) _Float16 Vt[Dh][VTP];   // [d][s]
  __shared__ alignas(16) _Float16 Ps[128][PSP];  // [t_local][s]

  const int h    = blockIdx.x;
  const int g    = h >> 2;          // n_rep = 4
  const int tid  = threadIdx.x;
  const int wave = tid >> 5;
  const int lane = tid & 31;
  const int r = lane & 15, hf = lane >> 4;
  const int pos   = *posp;
  const int total = pos + T_;
  const float scale = 0.08838834764831845f;  // 1/sqrt(128)

  // Persistent Q fragments for this wave's 16 query rows (f16 in HBM).
  const int trow = blockIdx.y * 128 + wave * 16;
  const _Float16* qbase = Qb + ((size_t)h * T_ + trow) * Dh;
  v16h qf[4];
#pragma unroll
  for (int i = 0; i < 4; ++i) qf[i] = ld_fragA(qbase, i * 32, Dh);

  v8f acc[8] = {};     // output accumulators: N = d = 16n + r
  float m[8], l[8];
#pragma unroll
  for (int j = 0; j < 8; ++j) { m[j] = -1e30f; l[j] = 0.f; }

  for (int s0 = 0; s0 < total; s0 += SC) {
    // ---- stage K chunk (64 x 128 fp32 -> f16 row-major) ----
    {
      const int row   = tid >> 2;
      const int dbase = (tid & 3) * 32;
      const int sg    = s0 + row;
      const float* src = (sg < pos)
          ? (kcache + ((size_t)g * MAXSEQ_ + sg) * Dh + dbase)
          : (Kext   + ((size_t)g * T_ + (sg - pos)) * Dh + dbase);
      unsigned tk[16];
      pack32(src, tk);
      store32(&Ks[row][dbase], tk);
      // prefetch next chunk's K row
      const int sn = sg + SC;
      if (sn < total) {
        const float* nsrc = (sn < pos)
            ? (kcache + ((size_t)g * MAXSEQ_ + sn) * Dh + dbase)
            : (Kext   + ((size_t)g * T_ + (sn - pos)) * Dh + dbase);
        __builtin_prefetch(nsrc, 0, 1);
      }
    }
    // ---- stage V chunk transposed (Vt[d][s], f16) ----
    {
      const int d0 = (tid & 7) * 16;
      const int sl = (tid >> 3) * 2;
      _Float16 hv[2][16];
#pragma unroll
      for (int ss = 0; ss < 2; ++ss) {
        const int sg = s0 + sl + ss;
        const float* src = (sg < pos)
            ? (vcache + ((size_t)g * MAXSEQ_ + sg) * Dh + d0)
            : (Vext   + ((size_t)g * T_ + (sg - pos)) * Dh + d0);
#pragma unroll
        for (int i = 0; i < 4; ++i) {
          v4f f = *(const v4f*)(src + i * 4);
          hv[ss][4*i+0] = (_Float16)f[0];
          hv[ss][4*i+1] = (_Float16)f[1];
          hv[ss][4*i+2] = (_Float16)f[2];
          hv[ss][4*i+3] = (_Float16)f[3];
        }
        const int sn = sg + SC;
        if (sn < total) {
          const float* nsrc = (sn < pos)
              ? (vcache + ((size_t)g * MAXSEQ_ + sn) * Dh + d0)
              : (Vext   + ((size_t)g * T_ + (sn - pos)) * Dh + d0);
          __builtin_prefetch(nsrc, 0, 1);
        }
      }
#pragma unroll
      for (int dd = 0; dd < 16; ++dd) {
        union { _Float16 h2[2]; unsigned u; } p;
        p.h2[0] = hv[0][dd]; p.h2[1] = hv[1][dd];
        *(unsigned*)(&Vt[d0 + dd][sl]) = p.u;
      }
    }
    __syncthreads();

    // ---- scores: Q(16x128) @ K_chunk^T -> 16x64 ----
    v8f sc[4] = {};
#pragma unroll
    for (int kk = 0; kk < 4; ++kk)
#pragma unroll
      for (int n = 0; n < 4; ++n) {
        v16h b = ld_fragB(&Ks[n * 16][0], kk * 32, KSP);
        sc[n] = wmma16(qf[kk], b, sc[n]);
      }
    // QK pipeline: 32 DS reads, 16 WMMAs (prime 4, steady {1,2} x14, drain 2).
    SGB(0x100, 4, 0);
#pragma unroll
    for (int i = 0; i < 14; ++i) { SGB(0x008, 1, 0); SGB(0x100, 2, 0); }
    SGB(0x008, 1, 0);
    SGB(0x008, 1, 0);

    // ---- online softmax (rows split: lanes<16 hold M=j, lanes>=16 M=j+8) ----
    float cmax[8];
#pragma unroll
    for (int j = 0; j < 8; ++j) {
      float v = -1e30f;
#pragma unroll
      for (int n = 0; n < 4; ++n) v = fmaxf(v, sc[n][j] * scale);
#pragma unroll
      for (int off = 8; off >= 1; off >>= 1)
        v = fmaxf(v, __shfl_xor(v, off, 16));
      cmax[j] = v;
    }
    float alpha[8], rsum[8];
#pragma unroll
    for (int j = 0; j < 8; ++j) {
      float mn = fmaxf(m[j], cmax[j]);
      alpha[j] = __expf(m[j] - mn);
      m[j] = mn;
      float sr = 0.f;
#pragma unroll
      for (int n = 0; n < 4; ++n) {
        float p = __expf(sc[n][j] * scale - mn);
        sc[n][j] = p;
        sr += p;
      }
#pragma unroll
      for (int off = 8; off >= 1; off >>= 1)
        sr += __shfl_xor(sr, off, 16);
      rsum[j] = sr;
    }
#pragma unroll
    for (int j = 0; j < 8; ++j) {
      l[j] = l[j] * alpha[j] + rsum[j];
#pragma unroll
      for (int n = 0; n < 8; ++n) acc[n][j] *= alpha[j];
    }

    // ---- P -> LDS (wave-private strip), then out += P @ V ----
#pragma unroll
    for (int n = 0; n < 4; ++n)
#pragma unroll
      for (int j = 0; j < 8; ++j)
        Ps[wave * 16 + hf * 8 + j][n * 16 + r] = (_Float16)sc[n][j];

    // Hoist both P A-fragments; then a uniform stream of 16 B-frag loads.
    v16h p0 = ld_fragA(&Ps[wave * 16][0], 0,  PSP);
    v16h p1 = ld_fragA(&Ps[wave * 16][0], 32, PSP);
#pragma unroll
    for (int n = 0; n < 8; ++n) {
      v16h b = ld_fragB(&Vt[n * 16][0], 0, VTP);
      acc[n] = wmma16(p0, b, acc[n]);
    }
#pragma unroll
    for (int n = 0; n < 8; ++n) {
      v16h b = ld_fragB(&Vt[n * 16][0], 32, VTP);
      acc[n] = wmma16(p1, b, acc[n]);
    }
    // PV pipeline: 36 DS reads, 16 WMMAs (prime 8, steady {1,2} x14, drain 2).
    SGB(0x100, 8, 0);
#pragma unroll
    for (int i = 0; i < 14; ++i) { SGB(0x008, 1, 0); SGB(0x100, 2, 0); }
    SGB(0x008, 1, 0);
    SGB(0x008, 1, 0);

    __syncthreads();  // protect Ks/Vt before next chunk's staging
  }

  // ---- normalize and store O[t][h*128 + d] (fp32) ----
  const int orow = trow + hf * 8;
#pragma unroll
  for (int j = 0; j < 8; ++j) {
    float inv = 1.f / l[j];
#pragma unroll
    for (int n = 0; n < 8; ++n)
      O[(size_t)(orow + j) * (H_ * Dh) + (size_t)h * Dh + n * 16 + r] =
          acc[n][j] * inv;
  }
}

// ---------------------------------------------------------------------------
extern "C" void kernel_launch(void* const* d_in, const int* in_sizes, int n_in,
                              void* d_out, int out_size, void* d_ws, size_t ws_size,
                              hipStream_t stream)
{
  (void)in_sizes; (void)n_in; (void)out_size; (void)ws_size;

  const float* x      = (const float*)d_in[0];
  const float* cosv   = (const float*)d_in[1];
  const float* sinv   = (const float*)d_in[2];
  const float* wq     = (const float*)d_in[3];
  const float* wk     = (const float*)d_in[4];
  const float* wv     = (const float*)d_in[5];
  const float* wo     = (const float*)d_in[6];
  const float* kcache = (const float*)d_in[7];
  const float* vcache = (const float*)d_in[8];
  const int*   posp   = (const int*)d_in[9];

  char* ws = (char*)d_ws;
  float*    Qraw = (float*)(ws);                         // 256*4096 f32  (4 MiB)
  float*    Kraw = (float*)(ws + (size_t)(4u << 20));    // 256*1024 f32  (1 MiB)
  float*    Vraw = (float*)(ws + (size_t)(5u << 20));    // 256*1024 f32  (1 MiB)
  _Float16* Qb   = (_Float16*)(ws + (size_t)(6u << 20)); // 32*256*128 f16 (2 MiB)
  float*    Kext = (float*)(ws + (size_t)(8u << 20));    // 8*256*128 f32 (1 MiB)
  float*    Vext = (float*)(ws + (size_t)(9u << 20));    // 8*256*128 f32 (1 MiB)
  float*    O    = (float*)(ws + (size_t)(10u << 20));   // 256*4096 f32  (4 MiB)

  dim3 blk(256);
  gemm_nt_kernel<<<dim3(DIM_ / BN, T_ / BM), blk, 0, stream>>>(x, wq, Qraw, DIM_, DIM_);
  gemm_nt_kernel<<<dim3((KVH_ * Dh) / BN, T_ / BM), blk, 0, stream>>>(x, wk, Kraw, KVH_ * Dh, DIM_);
  gemm_nt_kernel<<<dim3((KVH_ * Dh) / BN, T_ / BM), blk, 0, stream>>>(x, wv, Vraw, KVH_ * Dh, DIM_);
  rope_kernel<<<dim3(T_), blk, 0, stream>>>(Qraw, Kraw, Vraw, cosv, sinv, Qb, Kext, Vext);
  attn_kernel<<<dim3(H_, T_ / 128), blk, 0, stream>>>(Qb, kcache, vcache, Kext, Vext, posp, O);
  gemm_nt_kernel<<<dim3(DIM_ / BN, T_ / BM), blk, 0, stream>>>(O, wo, (float*)d_out, DIM_, DIM_);
}
